// SphericalChebBNPool_39084202394173
// MI455X (gfx1250) — matrix operand order repaired and checked by reading, not compiled, not run on hardware
//
#include <hip/hip_runtime.h>

#define VV    49152
#define BB    8
#define EE    393216
#define FIN   32
#define FOUT  64
#define VP    12288                 // V/4
#define NROWS (BB*VV)               // 393216 rows of the GEMM
#define MTILES (NROWS/16)           // 24576
#define NXF   (BB*VV*FIN)           // floats per Chebyshev buffer (12,582,912)
#define NYF   (NROWS*FOUT)          // floats in y (25,165,824)

typedef __attribute__((ext_vector_type(16))) _Float16 v16h;
typedef __attribute__((ext_vector_type(8)))  float    v8f;

__device__ __forceinline__ void atomAddF(float* p, float v) {
  unsafeAtomicAdd(p, v);            // native global_atomic_add_f32 on gfx1250
}

// ---- init: x1 = 0, x2 = -x0, zero BN stats ----------------------------------
__global__ __launch_bounds__(256) void k_init1(const float* __restrict__ x0,
                                               float* __restrict__ x1,
                                               float* __restrict__ x2,
                                               float* __restrict__ stats) {
  const size_t i = (size_t)blockIdx.x * 256 + threadIdx.x;
  if (i < (size_t)NXF) { x1[i] = 0.0f; x2[i] = -x0[i]; }
  if (blockIdx.x == 0 && threadIdx.x < 256) stats[threadIdx.x] = 0.0f;
}

// ---- init: x3 = -x1 (after x1 is complete) ----------------------------------
__global__ __launch_bounds__(256) void k_init2(const float* __restrict__ x1,
                                               float* __restrict__ x3) {
  const size_t i = (size_t)blockIdx.x * 256 + threadIdx.x;
  if (i < (size_t)NXF) x3[i] = -x1[i];
}

// ---- SpMM scatter: yout[b,row,:] += scale*vals[e]*xin[b,col,:] --------------
__global__ __launch_bounds__(256) void k_spmm(const float* __restrict__ xin,
                                              float* __restrict__ yout,
                                              const float* __restrict__ vals,
                                              const int* __restrict__ rows,
                                              const int* __restrict__ cols,
                                              float scale) {
  const int e    = blockIdx.x * 8 + (threadIdx.x >> 5);   // one wave per edge
  const int lane = threadIdx.x & 31;                      // lane = feature
  const float v  = vals[e] * scale;
  const size_t soff = (size_t)cols[e] * FIN + lane;
  const size_t doff = (size_t)rows[e] * FIN + lane;
#pragma unroll
  for (int b = 0; b < BB; ++b) {
    const size_t bo = (size_t)b * VV * FIN;
    atomAddF(yout + bo + doff, v * xin[bo + soff]);
  }
}

// ---- GEMM via v_wmma_f32_16x16x32_f16 + bias + BN partial stats -------------
__global__ __launch_bounds__(256) void k_gemm_wmma(
    const float* __restrict__ x0, const float* __restrict__ x1,
    const float* __restrict__ x2, const float* __restrict__ x3,
    const float* __restrict__ weight, const float* __restrict__ bias,
    float* __restrict__ y, float* __restrict__ stat) {
  const int lane  = threadIdx.x & 31;
  const int wave  = threadIdx.x >> 5;
  const int ntile = wave & 3;           // Fout tile (4 tiles of 16)
  const int hi    = lane >> 4;          // lane half
  const int l16   = lane & 15;
  const int n     = ntile * 16 + l16;   // output channel owned by this lane

  __shared__ float s_sum[FOUT];
  __shared__ float s_sq[FOUT];
  if (threadIdx.x < FOUT) { s_sum[threadIdx.x] = 0.0f; s_sq[threadIdx.x] = 0.0f; }
  __syncthreads();

  // B fragments per Chebyshev chunk; B layout (16-bit, 32x16):
  // lanes 0-15 -> K {0..7}U{16..23} of column n, lanes 16-31 -> K {8..15}U{24..31}.
  v16h bfrag[4];
#pragma unroll
  for (int c = 0; c < 4; ++c) {
    const float* w = weight + c * (FIN * FOUT) + (size_t)(hi * 8) * FOUT + n;
    v16h bf;
#pragma unroll
    for (int j = 0; j < 16; ++j) bf[j] = (_Float16)w[(j + (j & 8)) * FOUT];
    bfrag[c] = bf;
  }

  const float* xs[4] = {x0, x1, x2, x3};
  const float bval = bias[n];
  float lsum = 0.0f, lsq = 0.0f;

  const int mstart  = blockIdx.x * 2 + (wave >> 2);
  const int mstride = gridDim.x * 2;
  for (int mt = mstart; mt < MTILES; mt += mstride) {
    const size_t rbase = (size_t)mt * 16;
    v8f acc = {};
#pragma unroll
    for (int c = 0; c < 4; ++c) {
      // A layout (16-bit, 16x32): lane<16 -> M=l16, K {0..7}U{16..23};
      // lane>=16 -> M=l16, K {8..15}U{24..31}.
      const float* xr = xs[c] + (rbase + l16) * FIN + hi * 8;
      const float4 a0 = *(const float4*)(xr + 0);
      const float4 a1 = *(const float4*)(xr + 4);
      const float4 a2 = *(const float4*)(xr + 16);
      const float4 a3 = *(const float4*)(xr + 20);
      v16h af;
      af[0]=(_Float16)a0.x; af[1]=(_Float16)a0.y; af[2]=(_Float16)a0.z; af[3]=(_Float16)a0.w;
      af[4]=(_Float16)a1.x; af[5]=(_Float16)a1.y; af[6]=(_Float16)a1.z; af[7]=(_Float16)a1.w;
      af[8]=(_Float16)a2.x; af[9]=(_Float16)a2.y; af[10]=(_Float16)a2.z; af[11]=(_Float16)a2.w;
      af[12]=(_Float16)a3.x; af[13]=(_Float16)a3.y; af[14]=(_Float16)a3.z; af[15]=(_Float16)a3.w;
      acc = __builtin_amdgcn_wmma_f32_16x16x32_f16(
          /*neg_a=*/false, af, /*neg_b=*/false, bfrag[c],
          /*c_mod=*/(short)0, acc, /*reuse_a=*/false, /*reuse_b=*/false);
    }
    // C/D layout: VGPR j -> row (j + 8*hi), column n.
#pragma unroll
    for (int j = 0; j < 8; ++j) {
      const float vv = acc[j] + bval;
      y[(rbase + (size_t)hi * 8 + j) * FOUT + n] = vv;
      lsum += vv; lsq += vv * vv;
    }
    if (mt + mstride < MTILES)
      __builtin_prefetch(xs[0] + (size_t)(mt + mstride) * 16 * FIN, 0, 1);
  }

  atomicAdd(&s_sum[n], lsum);   // LDS ds_add_f32
  atomicAdd(&s_sq[n],  lsq);
  __syncthreads();
  if (threadIdx.x < FOUT) {
    atomAddF(&stat[threadIdx.x],        s_sum[threadIdx.x]);
    atomAddF(&stat[FOUT + threadIdx.x], s_sq[threadIdx.x]);
  }
}

// ---- BN finalize: per-channel affine a*x + b --------------------------------
__global__ void k_bn_finalize(const float* __restrict__ stat,
                              const float* __restrict__ gamma,
                              const float* __restrict__ beta,
                              float* __restrict__ ab) {
  const int c = threadIdx.x;
  if (c >= FOUT) return;
  const float inv  = 1.0f / (float)NROWS;
  const float mean = stat[c] * inv;
  const float var  = stat[FOUT + c] * inv - mean * mean;   // biased variance
  const float a    = gamma[c] * rsqrtf(var + 1e-5f);
  ab[c]        = a;
  ab[FOUT + c] = beta[c] - mean * a;
}

// ---- fused BN + ReLU + HEALPix max-pool(4) ----------------------------------
__global__ __launch_bounds__(256) void k_bn_relu_pool(const float* __restrict__ y,
                                                      const float* __restrict__ ab,
                                                      float* __restrict__ out) {
  const size_t idx = (size_t)blockIdx.x * 256 + threadIdx.x;
  if (idx >= (size_t)BB * VP * FOUT) return;
  const int c  = (int)(idx & 63);
  const int vp = (int)((idx >> 6) % VP);
  const int b  = (int)(idx / ((size_t)VP * FOUT));
  const float a = ab[c], sh = ab[FOUT + c];
  const size_t base = ((size_t)b * VV + (size_t)vp * 4) * FOUT + c;
  float m = 0.0f;                         // ReLU folds into max with 0 init
#pragma unroll
  for (int i = 0; i < 4; ++i) m = fmaxf(m, a * y[base + (size_t)i * FOUT] + sh);
  out[idx] = m;
}

extern "C" void kernel_launch(void* const* d_in, const int* in_sizes, int n_in,
                              void* d_out, int out_size, void* d_ws, size_t ws_size,
                              hipStream_t stream) {
  const float* x0     = (const float*)d_in[0];   // [8,49152,32]
  const float* vals   = (const float*)d_in[1];   // [E]
  const float* weight = (const float*)d_in[2];   // [4,32,64]
  const float* bias   = (const float*)d_in[3];   // [64]
  const float* gamma  = (const float*)d_in[4];   // [64]
  const float* beta   = (const float*)d_in[5];   // [64]
  const int*   rows   = (const int*)d_in[6];     // [E]
  const int*   cols   = (const int*)d_in[7];     // [E]

  float* ws   = (float*)d_ws;
  float* x1   = ws;                 // [B*V*32]
  float* x2   = x1 + NXF;           // [B*V*32]
  float* x3   = x2 + NXF;           // [B*V*32]
  float* y    = x3 + NXF;           // [B*V, 64]
  float* stat = y + NYF;            // sum[64], sumsq[64]
  float* ab   = stat + 128;         // scale[64], shift[64]

  const int initBlocks = (NXF + 255) / 256;
  k_init1<<<initBlocks, 256, 0, stream>>>(x0, x1, x2, stat);
  k_spmm <<<EE / 8, 256, 0, stream>>>(x0, x1, vals, rows, cols, 1.0f);   // x1 = L x0
  k_init2<<<initBlocks, 256, 0, stream>>>(x1, x3);                       // x3 = -x1
  k_spmm <<<EE / 8, 256, 0, stream>>>(x1, x2, vals, rows, cols, 2.0f);   // x2 = 2Lx1 - x0
  k_spmm <<<EE / 8, 256, 0, stream>>>(x2, x3, vals, rows, cols, 2.0f);   // x3 = 2Lx2 - x1
  k_gemm_wmma<<<3072, 256, 0, stream>>>(x0, x1, x2, x3, weight, bias, y, stat);
  k_bn_finalize<<<1, 64, 0, stream>>>(stat, gamma, beta, ab);
  k_bn_relu_pool<<<((size_t)BB * VP * FOUT + 255) / 256, 256, 0, stream>>>(y, ab, (float*)d_out);
}
